// SparseMoEBlock_39453569581632
// MI455X (gfx1250) — matrix operand order repaired and checked by reading, not compiled, not run on hardware
//
#include <hip/hip_runtime.h>
#include <hip/hip_bf16.h>

typedef __attribute__((ext_vector_type(16))) __bf16 v16bf;
typedef __attribute__((ext_vector_type(8)))  float  v8f;

#define T_TOK 4096
#define HDIM  1024
#define IDIM  2048
#define NEXP  8
#define BM    64          // token rows per block
#define BN    128         // output cols per block
#define KC    64          // K chunk per LDS stage (2 WMMA K-steps)
#define KU    (KC/2 + 1)  // LDS stride in dwords (odd -> conflict-free)

// ---- helpers ---------------------------------------------------------------
__device__ __forceinline__ unsigned pack2(float a, float b) {
  union { unsigned u; __bf16 h[2]; } cv;
  cv.h[0] = (__bf16)a; cv.h[1] = (__bf16)b;
  return cv.u;
}
__device__ __forceinline__ void set_pair(v16bf& f, int v, unsigned u) {
  union { unsigned u; __bf16 h[2]; } cv; cv.u = u;
  f[2*v] = cv.h[0]; f[2*v+1] = cv.h[1];
}
#define WMMA_BF16(A, B, C) \
  __builtin_amdgcn_wmma_f32_16x16x32_bf16(false, (A), false, (B), (short)0, (C), false, false)

// ---------------------------------------------------------------------------
// Router: one wave32 per token. logits = x . w_gate^T, top-2 softmax renorm,
// compact (token, gate) lists per expert via atomic append.
// ---------------------------------------------------------------------------
__global__ __launch_bounds__(256) void moe_router(
    const float* __restrict__ x, const float* __restrict__ w_gate,
    float* __restrict__ logits_out, int* __restrict__ e_count,
    int* __restrict__ e_tok, float* __restrict__ e_gate)
{
  const int wave = threadIdx.x >> 5;
  const int lane = threadIdx.x & 31;
  const int t = blockIdx.x * 8 + wave;
  if (t >= T_TOK) return;

  const float* __restrict__ xr = x + (size_t)t * HDIM;
  float lg[NEXP];
#pragma unroll
  for (int e = 0; e < NEXP; ++e) {
    const float* __restrict__ wr = w_gate + (size_t)e * HDIM;
    float acc = 0.f;
    for (int h = lane; h < HDIM; h += 32)
      acc = fmaf(xr[h], wr[h], acc);
#pragma unroll
    for (int off = 16; off > 0; off >>= 1)
      acc += __shfl_xor(acc, off, 32);
    lg[e] = acc;
  }

  if (lane == 0) {
    int e0 = 0;
#pragma unroll
    for (int e = 1; e < NEXP; ++e) if (lg[e] > lg[e0]) e0 = e;
    int e1 = (e0 == 0) ? 1 : 0;
#pragma unroll
    for (int e = 0; e < NEXP; ++e) if (e != e0 && lg[e] > lg[e1]) e1 = e;

    const float w0 = 1.f / (1.f + __expf(lg[e1] - lg[e0]));
    const float w1 = 1.f - w0;

#pragma unroll
    for (int e = 0; e < NEXP; ++e)
      logits_out[(size_t)t * NEXP + e] = lg[e];

    int p0 = atomicAdd(&e_count[e0], 1);
    e_tok[e0 * T_TOK + p0]  = t;
    e_gate[e0 * T_TOK + p0] = w0;
    int p1 = atomicAdd(&e_count[e1], 1);
    e_tok[e1 * T_TOK + p1]  = t;
    e_gate[e1 * T_TOK + p1] = w1;
  }
}

// ---------------------------------------------------------------------------
// GEMM1 + SwiGLU:  h[slot, i] = silu(x.wg^T) * (x.wu^T)  -> bf16 hbuf
// 256 thr = 8 waves as 2(M)x4(N); each wave owns a 32x32 output patch per
// GEMM (2x2 WMMA accumulators), 16 WMMAs per LDS stage per wave.
// ---------------------------------------------------------------------------
__global__ __launch_bounds__(256) void moe_gemm1(
    const float* __restrict__ x, const float* __restrict__ w_g,
    const float* __restrict__ w_u, const int* __restrict__ e_count,
    const int* __restrict__ e_tok, __bf16* __restrict__ hbuf)
{
  const int e     = blockIdx.z;
  const int count = e_count[e];
  const int row0  = blockIdx.x * BM;
  if (row0 >= count) return;            // dynamic sparsity: skip empty tiles
  const int i0 = blockIdx.y * BN;

  __shared__ unsigned xsu[BM][KU];      // x tile, bf16 pairs
  __shared__ unsigned gsu[BN][KU];      // w_g tile
  __shared__ unsigned usu[BN][KU];      // w_u tile
  __shared__ int      toks[BM];

  const int tid = threadIdx.x;
  if (tid < BM) {
    int slot = row0 + tid;
    if (slot >= count) slot = count - 1;     // padded rows: valid token, ignored later
    toks[tid] = e_tok[e * T_TOK + slot];
  }
  __syncthreads();

  const int lane = tid & 31;
  const int wv   = tid >> 5;
  const int wm   = wv & 1;     // rows 32*wm .. +31
  const int wn   = wv >> 1;    // cols 32*wn .. +31
  const int half = lane >> 4;
  const int mn   = lane & 15;

  v8f accg[2][2] = {{{}, {}}, {{}, {}}};
  v8f accu[2][2] = {{{}, {}}, {{}, {}}};

  const float* __restrict__ wg_base = w_g + ((size_t)e * IDIM + i0) * HDIM;
  const float* __restrict__ wu_base = w_u + ((size_t)e * IDIM + i0) * HDIM;

  for (int k0 = 0; k0 < HDIM; k0 += KC) {
    // ---- stage tiles: float4 global loads -> packed bf16 dwords in LDS ----
    for (int li = tid; li < BM * (KC/4); li += 256) {       // 1024 -> 4 iters
      const int r = li >> 4, c4 = li & 15;
      const float4 f = *(const float4*)(x + (size_t)toks[r] * HDIM + k0 + 4*c4);
      xsu[r][2*c4]   = pack2(f.x, f.y);
      xsu[r][2*c4+1] = pack2(f.z, f.w);
    }
    for (int li = tid; li < BN * (KC/4); li += 256) {       // 2048 -> 8 iters
      const int r = li >> 4, c4 = li & 15;
      const size_t off = (size_t)r * HDIM + k0 + 4*c4;
      if (k0 + KC < HDIM) {
        __builtin_prefetch(wg_base + off + KC, 0, 3);
        __builtin_prefetch(wu_base + off + KC, 0, 3);
      }
      const float4 fg = *(const float4*)(wg_base + off);
      const float4 fu = *(const float4*)(wu_base + off);
      gsu[r][2*c4]   = pack2(fg.x, fg.y);
      gsu[r][2*c4+1] = pack2(fg.z, fg.w);
      usu[r][2*c4]   = pack2(fu.x, fu.y);
      usu[r][2*c4+1] = pack2(fu.z, fu.w);
    }
    __syncthreads();

    // ---- 2 WMMA K-steps per stage --------------------------------------
#pragma unroll
    for (int ks = 0; ks < KC/32; ++ks) {
      v16bf a0, a1;
#pragma unroll
      for (int v = 0; v < 8; ++v) {
        const int ui = v + 8*half + 16*ks;   // A lane layout
        set_pair(a0, v, xsu[32*wm      + mn][ui]);
        set_pair(a1, v, xsu[32*wm + 16 + mn][ui]);
      }
#pragma unroll
      for (int nt = 0; nt < 2; ++nt) {
        const int colr = 32*wn + 16*nt + mn;
        v16bf bg, bu;
#pragma unroll
        for (int v = 0; v < 8; ++v) {
          const int ui = 8*half + v + 16*ks;                     // B lane layout
          set_pair(bg, v, gsu[colr][ui]);
          set_pair(bu, v, usu[colr][ui]);
        }
        accg[0][nt] = WMMA_BF16(a0, bg, accg[0][nt]);
        accg[1][nt] = WMMA_BF16(a1, bg, accg[1][nt]);
        accu[0][nt] = WMMA_BF16(a0, bu, accu[0][nt]);
        accu[1][nt] = WMMA_BF16(a1, bu, accu[1][nt]);
      }
    }
    __syncthreads();
  }

  // ---- epilogue: h = silu(g) * u -> bf16 hbuf ---------------------------
  const size_t hbase = ((size_t)e * T_TOK + row0) * IDIM;
#pragma unroll
  for (int mt = 0; mt < 2; ++mt) {
#pragma unroll
    for (int nt = 0; nt < 2; ++nt) {
      const int col = i0 + 32*wn + 16*nt + mn;
#pragma unroll
      for (int r = 0; r < 8; ++r) {
        const int m = 32*wm + 16*mt + r + 8*half;
        const float g = accg[mt][nt][r];
        const float u = accu[mt][nt][r];
        const float h = g * u * __builtin_amdgcn_rcpf(1.f + __expf(-g));
        hbuf[hbase + (size_t)m * IDIM + col] = (__bf16)h;
      }
    }
  }
}

// ---------------------------------------------------------------------------
// GEMM2 + weighted scatter:  out[tok, h] += gate * (h . w_d^T)
// ---------------------------------------------------------------------------
__global__ __launch_bounds__(256) void moe_gemm2(
    const __bf16* __restrict__ hbuf, const float* __restrict__ w_d,
    const int* __restrict__ e_count, const int* __restrict__ e_tok,
    const float* __restrict__ e_gate, float* __restrict__ out)
{
  const int e     = blockIdx.z;
  const int count = e_count[e];
  const int row0  = blockIdx.x * BM;
  if (row0 >= count) return;
  const int h0 = blockIdx.y * BN;

  __shared__ unsigned hsu[BM][KU];
  __shared__ unsigned dsu[BN][KU];
  __shared__ int      toks[BM];
  __shared__ float    gts[BM];

  const int tid = threadIdx.x;
  if (tid < BM) {
    const int slot   = row0 + tid;
    const bool valid = slot < count;
    const int cs = valid ? slot : (count - 1);
    toks[tid] = e_tok[e * T_TOK + cs];
    gts[tid]  = valid ? e_gate[e * T_TOK + cs] : 0.f;  // padded rows add +0.0
  }
  __syncthreads();

  const int lane = tid & 31;
  const int wv   = tid >> 5;
  const int wm   = wv & 1;
  const int wn   = wv >> 1;
  const int half = lane >> 4;
  const int mn   = lane & 15;

  v8f acc[2][2] = {{{}, {}}, {{}, {}}};

  const __bf16* __restrict__ hrow    = hbuf + ((size_t)e * T_TOK + row0) * IDIM;
  const float*  __restrict__ wd_base = w_d  + ((size_t)e * HDIM + h0) * IDIM;

  for (int k0 = 0; k0 < IDIM; k0 += KC) {
    // h tile already bf16: uint4 (b128) loads straight into LDS dwords
    for (int li = tid; li < BM * (KC/8); li += 256) {        // 512 -> 2 iters
      const int r = li >> 3, c8 = li & 7;
      const uint4 q = *(const uint4*)(hrow + (size_t)r * IDIM + k0 + 8*c8);
      hsu[r][4*c8]   = q.x;
      hsu[r][4*c8+1] = q.y;
      hsu[r][4*c8+2] = q.z;
      hsu[r][4*c8+3] = q.w;
    }
    for (int li = tid; li < BN * (KC/4); li += 256) {        // 2048 -> 8 iters
      const int r = li >> 4, c4 = li & 15;
      const size_t off = (size_t)r * IDIM + k0 + 4*c4;
      if (k0 + KC < IDIM) __builtin_prefetch(wd_base + off + KC, 0, 3);
      const float4 f = *(const float4*)(wd_base + off);
      dsu[r][2*c4]   = pack2(f.x, f.y);
      dsu[r][2*c4+1] = pack2(f.z, f.w);
    }
    __syncthreads();

#pragma unroll
    for (int ks = 0; ks < KC/32; ++ks) {
      v16bf a0, a1;
#pragma unroll
      for (int v = 0; v < 8; ++v) {
        const int ui = v + 8*half + 16*ks;
        set_pair(a0, v, hsu[32*wm      + mn][ui]);
        set_pair(a1, v, hsu[32*wm + 16 + mn][ui]);
      }
#pragma unroll
      for (int nt = 0; nt < 2; ++nt) {
        const int colr = 32*wn + 16*nt + mn;
        v16bf b;
#pragma unroll
        for (int v = 0; v < 8; ++v)
          set_pair(b, v, dsu[colr][8*half + v + 16*ks]);
        acc[0][nt] = WMMA_BF16(a0, b, acc[0][nt]);
        acc[1][nt] = WMMA_BF16(a1, b, acc[1][nt]);
      }
    }
    __syncthreads();
  }

  // ---- weighted scatter into out (exactly 2 adds per (t,h) -> deterministic)
#pragma unroll
  for (int mt = 0; mt < 2; ++mt) {
#pragma unroll
    for (int nt = 0; nt < 2; ++nt) {
      const int col = h0 + 32*wn + 16*nt + mn;
#pragma unroll
      for (int r = 0; r < 8; ++r) {
        const int m = 32*wm + 16*mt + r + 8*half;
        atomicAdd(&out[(size_t)toks[m] * HDIM + col], gts[m] * acc[mt][nt][r]);
      }
    }
  }
}

// ---------------------------------------------------------------------------
extern "C" void kernel_launch(void* const* d_in, const int* in_sizes, int n_in,
                              void* d_out, int out_size, void* d_ws, size_t ws_size,
                              hipStream_t stream) {
  (void)in_sizes; (void)n_in; (void)out_size; (void)ws_size;
  const float* x      = (const float*)d_in[0];   // [T,H]
  const float* w_gate = (const float*)d_in[1];   // [E,H]
  const float* w_g    = (const float*)d_in[2];   // [E,I,H]
  const float* w_u    = (const float*)d_in[3];   // [E,I,H]
  const float* w_d    = (const float*)d_in[4];   // [E,H,I]

  float* out    = (float*)d_out;                 // [T,H]
  float* logits = out + (size_t)T_TOK * HDIM;    // [T,E]

  char*   ws      = (char*)d_ws;
  int*    e_count = (int*)ws;                                           // E ints
  int*    e_tok   = (int*)(ws + 256);                                   // E*T ints
  float*  e_gate  = (float*)(ws + 256 + (size_t)NEXP * T_TOK * 4);      // E*T f32
  __bf16* hbuf    = (__bf16*)(ws + 256 + 2 * (size_t)NEXP * T_TOK * 4); // E*T*I bf16

  hipMemsetAsync(e_count, 0, NEXP * sizeof(int), stream);
  hipMemsetAsync(out, 0, (size_t)T_TOK * HDIM * sizeof(float), stream);

  moe_router<<<T_TOK / 8, 256, 0, stream>>>(x, w_gate, logits, e_count, e_tok, e_gate);
  moe_gemm1<<<dim3(T_TOK / BM, IDIM / BN, NEXP), 256, 0, stream>>>(
      x, w_g, w_u, e_count, e_tok, hbuf);
  moe_gemm2<<<dim3(T_TOK / BM, HDIM / BN, NEXP), 256, 0, stream>>>(
      hbuf, w_d, e_count, e_tok, e_gate, out);
}
